// REM_57939108823047
// MI455X (gfx1250) — matrix operand order repaired and checked by reading, not compiled, not run on hardware
//
#include <hip/hip_runtime.h>

// ---------------- types for gfx1250 WMMA ----------------
typedef __attribute__((ext_vector_type(16))) __bf16 v16bf;
typedef __attribute__((ext_vector_type(8)))  float  v8f;

union BFrag {                  // 8 VGPRs = 16 bf16
    v16bf v;
    unsigned int u[8];
};

__device__ __forceinline__ unsigned short bf16_1(float f) {
    unsigned int u = __float_as_uint(f);               // round-to-nearest-even
    return (unsigned short)((u + 0x7FFFu + ((u >> 16) & 1u)) >> 16);
}

__device__ __forceinline__ v8f wmma_bf16(const BFrag& a, const BFrag& b, v8f c) {
    return __builtin_amdgcn_wmma_f32_16x16x32_bf16(
        /*neg_a=*/false, a.v, /*neg_b=*/false, b.v,
        /*c_mod=*/(short)0, c, /*reuse_a=*/false, /*reuse_b=*/false);
}

// K-pair offset within a 32-chunk for A fragments (lane half `ah` = hi*8): contiguous k = 16*hi + 2*v
#define AOFF(v, ah) (((ah) << 1) + ((v) << 1))

// ---------------- prologue kernels (one-shot, memory-bound) ----------------
__global__ void k_cvt_bf16(const float* __restrict__ s, unsigned short* __restrict__ d, int n) {
    int i = blockIdx.x * blockDim.x + threadIdx.x;
    if (i < n) d[i] = bf16_1(s[i]);
}

// w2: OIHW [64][32][4][4]  ->  [n][(ky*4+kx)*32 + c]
__global__ void k_perm_w2(const float* __restrict__ w, unsigned short* __restrict__ d) {
    int t = blockIdx.x * blockDim.x + threadIdx.x;
    if (t >= 64 * 512) return;
    int n = t >> 9, k = t & 511;
    int s = k >> 5, c = k & 31, ky = s >> 2, kx = s & 3;
    d[t] = bf16_1(w[n * 512 + c * 16 + ky * 4 + kx]);
}

// w3: OIHW [64][64][3][3]  ->  [n][(ky*3+kx)*64 + c]
__global__ void k_perm_w3(const float* __restrict__ w, unsigned short* __restrict__ d) {
    int t = blockIdx.x * blockDim.x + threadIdx.x;
    if (t >= 64 * 576) return;
    int n = t / 576, k = t - n * 576;
    int s = k >> 6, c = k & 63;
    int ky = (s * 11) >> 5, kx = s - ky * 3;           // s<9: /3, %3 without div
    d[t] = bf16_1(w[n * 576 + c * 9 + ky * 3 + kx]);
}

// wl: [512][3136] indexed NCHW-flat (c*49+p)  ->  [n][p*64 + c]  (matches NHWC act3)
__global__ void k_perm_wl(const float* __restrict__ w, unsigned short* __restrict__ d) {
    int t = blockIdx.x * blockDim.x + threadIdx.x;
    if (t >= 512 * 3136) return;
    int n = t / 3136, f = t - n * 3136;
    int p = f >> 6, c = f & 63;
    d[t] = bf16_1(w[n * 3136 + c * 49 + p]);
}

// w_eff[a][d] = sum_h alphas[h] * wh[h][a][d]; padded to 32 rows of zeros
__global__ void k_heads(const float* __restrict__ alphas, const float* __restrict__ wh,
                        unsigned short* __restrict__ weffb) {
    int t = blockIdx.x * blockDim.x + threadIdx.x;
    if (t >= 32 * 512) return;
    int a = t >> 9, d = t & 511;
    float s = 0.f;
    if (a < 18)
        for (int h = 0; h < 200; ++h) s += alphas[h] * wh[(h * 18 + a) * 512 + d];
    weffb[t] = (a < 18) ? bf16_1(s) : (unsigned short)0;
}

__global__ void k_beff(const float* __restrict__ alphas, const float* __restrict__ bh,
                       float* __restrict__ beff) {
    int t = threadIdx.x;
    if (t >= 18) return;
    float s = 0.f;
    for (int h = 0; h < 200; ++h) s += alphas[h] * bh[h * 18 + t];
    beff[t] = s;
}

// ---------------- conv1: M=1024*400, N=32, K=256 (NCHW bf16 input), stride 4 ----------------
// one wave = 32(M) x 32(N) tile; output act1 NHWC [b][20][20][32] -> index m*32 + n
__global__ void k_conv1(const unsigned short* __restrict__ xb, const unsigned short* __restrict__ w1b,
                        const float* __restrict__ b1, unsigned short* __restrict__ act1) {
    const int wave = blockIdx.x * (blockDim.x >> 5) + (threadIdx.x >> 5);  // = mtile32
    const int l = threadIdx.x & 31, lm = l & 15, hi = l >> 4;
    const int ah = hi * 8, bh16 = hi * 16;
    // two A rows per lane: m0 and m0+16
    int m0 = wave * 32 + lm, m1 = m0 + 16;
    int b0 = m0 / 400, p0 = m0 - b0 * 400, oy0 = p0 / 20, ox0 = p0 - oy0 * 20;
    int b1i = m1 / 400, p1 = m1 - b1i * 400, oy1 = p1 / 20, ox1 = p1 - oy1 * 20;
    const unsigned short* xp0 = xb + (size_t)b0 * 28224 + (oy0 * 4) * 84 + ox0 * 4;
    const unsigned short* xp1 = xb + (size_t)b1i * 28224 + (oy1 * 4) * 84 + ox1 * 4;
    // two B cols per lane: n0 = lm, n1 = lm+16
    const unsigned short* wn0 = w1b + lm * 256;
    const unsigned short* wn1 = w1b + (lm + 16) * 256;
    v8f a00 = {}, a01 = {}, a10 = {}, a11 = {};
    for (int kc = 0; kc < 256; kc += 32) {
        BFrag A0, A1, B0, B1;
#pragma unroll
        for (int v = 0; v < 8; ++v) {
            int k = kc + AOFF(v, ah);
            int c = k >> 6, r = k & 63, ky = r >> 3, kx = r & 7;
            int ai = c * 7056 + ky * 84 + kx;
            A0.u[v] = *(const unsigned int*)(xp0 + ai);
            A1.u[v] = *(const unsigned int*)(xp1 + ai);
            int bi = kc + bh16 + (v << 1);
            B0.u[v] = *(const unsigned int*)(wn0 + bi);
            B1.u[v] = *(const unsigned int*)(wn1 + bi);
        }
        a00 = wmma_bf16(A0, B0, a00);
        a01 = wmma_bf16(A0, B1, a01);
        a10 = wmma_bf16(A1, B0, a10);
        a11 = wmma_bf16(A1, B1, a11);
    }
#pragma unroll
    for (int v = 0; v < 8; ++v) {
        int mo0 = wave * 32 + hi * 8 + v, mo1 = mo0 + 16;
        int no0 = lm, no1 = lm + 16;
        float s0 = b1[no0], s1 = b1[no1];
        float v00 = a00[v] + s0, v01 = a01[v] + s1, v10 = a10[v] + s0, v11 = a11[v] + s1;
        act1[(size_t)mo0 * 32 + no0] = bf16_1(v00 > 0.f ? v00 : 0.f);
        act1[(size_t)mo0 * 32 + no1] = bf16_1(v01 > 0.f ? v01 : 0.f);
        act1[(size_t)mo1 * 32 + no0] = bf16_1(v10 > 0.f ? v10 : 0.f);
        act1[(size_t)mo1 * 32 + no1] = bf16_1(v11 > 0.f ? v11 : 0.f);
    }
}

// ---------------- conv2: M=1024*81, N=64, K=512 ((ky*4+kx)*32+c), stride 2 ----------------
// input act1 NHWC [b][20][20][32]; output act2 NHWC [b][9][9][64]; 32x32 tile per wave
__global__ void k_conv2(const unsigned short* __restrict__ act1, const unsigned short* __restrict__ w2b,
                        const float* __restrict__ b2, unsigned short* __restrict__ act2) {
    const int wave = blockIdx.x * (blockDim.x >> 5) + (threadIdx.x >> 5);
    const int mt = wave >> 1, nt = wave & 1;            // NT = 64/32 = 2
    const int l = threadIdx.x & 31, lm = l & 15, hi = l >> 4;
    const int ah = hi * 8, bh16 = hi * 16;
    int m0 = mt * 32 + lm, m1 = m0 + 16;
    int b0 = m0 / 81, p0 = m0 - b0 * 81, oy0 = p0 / 9, ox0 = p0 - oy0 * 9;
    int b1i = m1 / 81, p1 = m1 - b1i * 81, oy1 = p1 / 9, ox1 = p1 - oy1 * 9;
    const unsigned short* ab0 = act1 + (size_t)b0 * 12800 + (oy0 * 2) * 640 + (ox0 * 2) * 32;
    const unsigned short* ab1 = act1 + (size_t)b1i * 12800 + (oy1 * 2) * 640 + (ox1 * 2) * 32;
    const unsigned short* wn0 = w2b + (nt * 32 + lm) * 512;
    const unsigned short* wn1 = w2b + (nt * 32 + lm + 16) * 512;
    v8f a00 = {}, a01 = {}, a10 = {}, a11 = {};
    for (int kc = 0; kc < 512; kc += 32) {
        const int s = kc >> 5, ky = s >> 2, kx = s & 3;  // chunk-constant spatial
        const int so = ky * 640 + kx * 32;
        BFrag A0, A1, B0, B1;
#pragma unroll
        for (int v = 0; v < 8; ++v) {
            int cv = AOFF(v, ah);                        // loop-invariant channel pair
            A0.u[v] = *(const unsigned int*)(ab0 + so + cv);
            A1.u[v] = *(const unsigned int*)(ab1 + so + cv);
            int bi = kc + bh16 + (v << 1);
            B0.u[v] = *(const unsigned int*)(wn0 + bi);
            B1.u[v] = *(const unsigned int*)(wn1 + bi);
        }
        a00 = wmma_bf16(A0, B0, a00);
        a01 = wmma_bf16(A0, B1, a01);
        a10 = wmma_bf16(A1, B0, a10);
        a11 = wmma_bf16(A1, B1, a11);
    }
#pragma unroll
    for (int v = 0; v < 8; ++v) {
        int mo0 = mt * 32 + hi * 8 + v, mo1 = mo0 + 16;
        int no0 = nt * 32 + lm, no1 = no0 + 16;
        float s0 = b2[no0], s1 = b2[no1];
        float v00 = a00[v] + s0, v01 = a01[v] + s1, v10 = a10[v] + s0, v11 = a11[v] + s1;
        act2[(size_t)mo0 * 64 + no0] = bf16_1(v00 > 0.f ? v00 : 0.f);
        act2[(size_t)mo0 * 64 + no1] = bf16_1(v01 > 0.f ? v01 : 0.f);
        act2[(size_t)mo1 * 64 + no0] = bf16_1(v10 > 0.f ? v10 : 0.f);
        act2[(size_t)mo1 * 64 + no1] = bf16_1(v11 > 0.f ? v11 : 0.f);
    }
}

// ---------------- conv3: M=1024*49, N=64, K=576 ((ky*3+kx)*64+c), stride 1 ----------------
// input act2 NHWC [b][9][9][64]; output act3 NHWC-flat [b][p][c]; 32x32 tile per wave
__global__ void k_conv3(const unsigned short* __restrict__ act2, const unsigned short* __restrict__ w3b,
                        const float* __restrict__ b3, unsigned short* __restrict__ act3) {
    const int wave = blockIdx.x * (blockDim.x >> 5) + (threadIdx.x >> 5);
    const int mt = wave >> 1, nt = wave & 1;            // NT = 2
    const int l = threadIdx.x & 31, lm = l & 15, hi = l >> 4;
    const int ah = hi * 8, bh16 = hi * 16;
    int m0 = mt * 32 + lm, m1 = m0 + 16;
    int b0 = m0 / 49, p0 = m0 - b0 * 49, oy0 = p0 / 7, ox0 = p0 - oy0 * 7;
    int b1i = m1 / 49, p1 = m1 - b1i * 49, oy1 = p1 / 7, ox1 = p1 - oy1 * 7;
    const unsigned short* ab0 = act2 + (size_t)b0 * 5184 + oy0 * 576 + ox0 * 64;
    const unsigned short* ab1 = act2 + (size_t)b1i * 5184 + oy1 * 576 + ox1 * 64;
    const unsigned short* wn0 = w3b + (nt * 32 + lm) * 576;
    const unsigned short* wn1 = w3b + (nt * 32 + lm + 16) * 576;
    v8f a00 = {}, a01 = {}, a10 = {}, a11 = {};
    for (int kc = 0; kc < 576; kc += 32) {
        const int s  = kc >> 6;                          // chunk-constant spatial (0..8)
        const int ky = (s * 11) >> 5, kx = s - ky * 3;
        const int so = ky * 576 + kx * 64 + (kc & 32);
        BFrag A0, A1, B0, B1;
#pragma unroll
        for (int v = 0; v < 8; ++v) {
            int cv = AOFF(v, ah);
            A0.u[v] = *(const unsigned int*)(ab0 + so + cv);
            A1.u[v] = *(const unsigned int*)(ab1 + so + cv);
            int bi = kc + bh16 + (v << 1);
            B0.u[v] = *(const unsigned int*)(wn0 + bi);
            B1.u[v] = *(const unsigned int*)(wn1 + bi);
        }
        a00 = wmma_bf16(A0, B0, a00);
        a01 = wmma_bf16(A0, B1, a01);
        a10 = wmma_bf16(A1, B0, a10);
        a11 = wmma_bf16(A1, B1, a11);
    }
#pragma unroll
    for (int v = 0; v < 8; ++v) {
        int mo0 = mt * 32 + hi * 8 + v, mo1 = mo0 + 16;
        int no0 = nt * 32 + lm, no1 = no0 + 16;
        float s0 = b3[no0], s1 = b3[no1];
        float v00 = a00[v] + s0, v01 = a01[v] + s1, v10 = a10[v] + s0, v11 = a11[v] + s1;
        act3[(size_t)mo0 * 64 + no0] = bf16_1(v00 > 0.f ? v00 : 0.f);
        act3[(size_t)mo0 * 64 + no1] = bf16_1(v01 > 0.f ? v01 : 0.f);
        act3[(size_t)mo1 * 64 + no0] = bf16_1(v10 > 0.f ? v10 : 0.f);
        act3[(size_t)mo1 * 64 + no1] = bf16_1(v11 > 0.f ? v11 : 0.f);
    }
}

// ---------------- FC: M=1024, N=512, K=3136 (contiguous); 32x32 tile per wave ----------------
__global__ void k_linear(const unsigned short* __restrict__ act3, const unsigned short* __restrict__ wlb,
                         const float* __restrict__ bl, unsigned short* __restrict__ h2) {
    const int wave = blockIdx.x * (blockDim.x >> 5) + (threadIdx.x >> 5);
    const int mt = wave >> 4, nt = wave & 15;           // NT = 512/32 = 16
    const int l = threadIdx.x & 31, lm = l & 15, hi = l >> 4;
    const int ah = hi * 8, bh16 = hi * 16;
    const unsigned short* am0 = act3 + (size_t)(mt * 32 + lm) * 3136;
    const unsigned short* am1 = am0 + (size_t)16 * 3136;
    const unsigned short* wn0 = wlb + (size_t)(nt * 32 + lm) * 3136;
    const unsigned short* wn1 = wn0 + (size_t)16 * 3136;
    v8f a00 = {}, a01 = {}, a10 = {}, a11 = {};
    for (int kc = 0; kc < 3136; kc += 32) {
        BFrag A0, A1, B0, B1;
#pragma unroll
        for (int v = 0; v < 8; ++v) {
            int k = kc + AOFF(v, ah);
            A0.u[v] = *(const unsigned int*)(am0 + k);
            A1.u[v] = *(const unsigned int*)(am1 + k);
            int bi = kc + bh16 + (v << 1);
            B0.u[v] = *(const unsigned int*)(wn0 + bi);
            B1.u[v] = *(const unsigned int*)(wn1 + bi);
        }
        a00 = wmma_bf16(A0, B0, a00);
        a01 = wmma_bf16(A0, B1, a01);
        a10 = wmma_bf16(A1, B0, a10);
        a11 = wmma_bf16(A1, B1, a11);
    }
#pragma unroll
    for (int v = 0; v < 8; ++v) {
        int mo0 = mt * 32 + hi * 8 + v, mo1 = mo0 + 16;
        int no0 = nt * 32 + lm, no1 = no0 + 16;
        float s0 = bl[no0], s1 = bl[no1];
        float v00 = a00[v] + s0, v01 = a01[v] + s1, v10 = a10[v] + s0, v11 = a11[v] + s1;
        h2[(size_t)mo0 * 512 + no0] = bf16_1(v00 > 0.f ? v00 : 0.f);
        h2[(size_t)mo0 * 512 + no1] = bf16_1(v01 > 0.f ? v01 : 0.f);
        h2[(size_t)mo1 * 512 + no0] = bf16_1(v10 > 0.f ? v10 : 0.f);
        h2[(size_t)mo1 * 512 + no1] = bf16_1(v11 > 0.f ? v11 : 0.f);
    }
}

// ---------------- final: M=1024, N=18 (weights zero-padded to 32), K=512; 32x32 tile ----------------
__global__ void k_final(const unsigned short* __restrict__ h2, const unsigned short* __restrict__ weffb,
                        const float* __restrict__ beff, float* __restrict__ out) {
    const int wave = blockIdx.x * (blockDim.x >> 5) + (threadIdx.x >> 5);  // = mtile32
    const int l = threadIdx.x & 31, lm = l & 15, hi = l >> 4;
    const int ah = hi * 8, bh16 = hi * 16;
    const unsigned short* am0 = h2 + (size_t)(wave * 32 + lm) * 512;
    const unsigned short* am1 = am0 + (size_t)16 * 512;
    const unsigned short* wn0 = weffb + (size_t)lm * 512;        // cols 0..15
    const unsigned short* wn1 = weffb + (size_t)(lm + 16) * 512; // cols 16..31 (18+ zero)
    v8f a00 = {}, a01 = {}, a10 = {}, a11 = {};
    for (int kc = 0; kc < 512; kc += 32) {
        BFrag A0, A1, B0, B1;
#pragma unroll
        for (int v = 0; v < 8; ++v) {
            int k = kc + AOFF(v, ah);
            A0.u[v] = *(const unsigned int*)(am0 + k);
            A1.u[v] = *(const unsigned int*)(am1 + k);
            int bi = kc + bh16 + (v << 1);
            B0.u[v] = *(const unsigned int*)(wn0 + bi);
            B1.u[v] = *(const unsigned int*)(wn1 + bi);
        }
        a00 = wmma_bf16(A0, B0, a00);
        a01 = wmma_bf16(A0, B1, a01);
        a10 = wmma_bf16(A1, B0, a10);
        a11 = wmma_bf16(A1, B1, a11);
    }
#pragma unroll
    for (int v = 0; v < 8; ++v) {
        int mo0 = wave * 32 + hi * 8 + v, mo1 = mo0 + 16;
        int no0 = lm, no1 = lm + 16;
        out[mo0 * 18 + no0] = a00[v] + beff[no0];        // no0 < 16 always valid
        out[mo1 * 18 + no0] = a10[v] + beff[no0];
        if (no1 < 18) {
            out[mo0 * 18 + no1] = a01[v] + beff[no1];
            out[mo1 * 18 + no1] = a11[v] + beff[no1];
        }
    }
}

// ---------------- host launcher ----------------
extern "C" void kernel_launch(void* const* d_in, const int* in_sizes, int n_in,
                              void* d_out, int out_size, void* d_ws, size_t ws_size,
                              hipStream_t stream) {
    const float* x      = (const float*)d_in[0];
    const float* alphas = (const float*)d_in[1];
    const float* w1 = (const float*)d_in[2];
    const float* b1 = (const float*)d_in[3];
    const float* w2 = (const float*)d_in[4];
    const float* b2 = (const float*)d_in[5];
    const float* w3 = (const float*)d_in[6];
    const float* b3 = (const float*)d_in[7];
    const float* wl = (const float*)d_in[8];
    const float* bl = (const float*)d_in[9];
    const float* wh = (const float*)d_in[10];
    const float* bh = (const float*)d_in[11];
    float* out = (float*)d_out;

    char* ws = (char*)d_ws;
    size_t off = 0;
    auto alloc = [&](size_t bytes) -> void* {
        void* p = (void*)(ws + off);
        off += (bytes + 255) & ~(size_t)255;
        return p;
    };
    unsigned short* xb    = (unsigned short*)alloc((size_t)1024 * 4 * 84 * 84 * 2); // 57.8 MB
    unsigned short* w1b   = (unsigned short*)alloc((size_t)32 * 256 * 2);
    unsigned short* w2b   = (unsigned short*)alloc((size_t)64 * 512 * 2);
    unsigned short* w3b   = (unsigned short*)alloc((size_t)64 * 576 * 2);
    unsigned short* wlb   = (unsigned short*)alloc((size_t)512 * 3136 * 2);
    unsigned short* weffb = (unsigned short*)alloc((size_t)32 * 512 * 2);
    float*          beff  = (float*)alloc(18 * 4);
    unsigned short* act1  = (unsigned short*)alloc((size_t)1024 * 400 * 32 * 2);    // NHWC
    unsigned short* act2  = (unsigned short*)alloc((size_t)1024 * 81 * 64 * 2);     // NHWC
    unsigned short* act3  = (unsigned short*)alloc((size_t)1024 * 49 * 64 * 2);     // NHWC
    unsigned short* h2    = (unsigned short*)alloc((size_t)1024 * 512 * 2);
    if (off > ws_size) return;   // workspace too small: do nothing (deterministic)

    const int NX = 1024 * 4 * 84 * 84;
    k_cvt_bf16<<<(NX + 255) / 256, 256, 0, stream>>>(x, xb, NX);
    k_cvt_bf16<<<(32 * 256 + 255) / 256, 256, 0, stream>>>(w1, w1b, 32 * 256);
    k_perm_w2<<<(64 * 512 + 255) / 256, 256, 0, stream>>>(w2, w2b);
    k_perm_w3<<<(64 * 576 + 255) / 256, 256, 0, stream>>>(w3, w3b);
    k_perm_wl<<<(512 * 3136 + 255) / 256, 256, 0, stream>>>(wl, wlb);
    k_heads<<<(32 * 512 + 255) / 256, 256, 0, stream>>>(alphas, wh, weffb);
    k_beff<<<1, 32, 0, stream>>>(alphas, bh, beff);

    // pipeline: 4 waves (128 thr) per block, one 32x32 output tile per wave (4 WMMA accs)
    k_conv1 <<<3200, 128, 0, stream>>>(xb, w1b, b1, act1);    // 12800 waves (M/32, N=32)
    k_conv2 <<<1296, 128, 0, stream>>>(act1, w2b, b2, act2);  //  5184 waves (2592 * 2)
    k_conv3 <<< 784, 128, 0, stream>>>(act2, w3b, b3, act3);  //  3136 waves (1568 * 2)
    k_linear<<< 128, 128, 0, stream>>>(act3, wlb, bl, h2);    //   512 waves (32 * 16)
    k_final <<<   8, 128, 0, stream>>>(h2, weffb, beff, out); //    32 waves
}
